// LDGCNN_70617852281554
// MI455X (gfx1250) — compile-verified, matrix operand order checked by reading, not run on hardware
//
#include <hip/hip_runtime.h>
#include <hip/hip_bf16.h>
#include <float.h>

#define KNN   20
#define NPTS  2048
#define NB    16
#define SLOPE 0.2f
#define NT    4      // N-tiles (16 cols each) per wave -> 16x64 output tile

typedef __attribute__((ext_vector_type(16))) __bf16 v16bf;
typedef __attribute__((ext_vector_type(8)))  float  v8f;

__device__ __forceinline__ unsigned short f2bf(float f) {
  union { float f; unsigned u; } v; v.f = f;
  unsigned r = v.u + 0x7FFFu + ((v.u >> 16) & 1u);   // round-to-nearest-even
  return (unsigned short)(r >> 16);
}

__device__ __forceinline__ float lrelu(float v) { return fmaxf(v, SLOPE * v); }

// ---------------------------------------------------------------------------
// 1) kNN: one block per 256 points of one batch; batch points staged in LDS.
//    Stable top-20 (ascending dist, strict <) matches jax.lax.top_k ordering.
// ---------------------------------------------------------------------------
__global__ void knn_kernel(const float* __restrict__ x, int* __restrict__ idx) {
  __shared__ float sx[NPTS], sy[NPTS], sz[NPTS];
  int b = blockIdx.x >> 3;          // 8 blocks per batch
  int chunk = blockIdx.x & 7;
  const float* xb = x + (size_t)b * NPTS * 3;
  for (int i = threadIdx.x; i < NPTS; i += 256) {
    sx[i] = xb[i * 3 + 0];
    sy[i] = xb[i * 3 + 1];
    sz[i] = xb[i * 3 + 2];
  }
  __syncthreads();

  int n = chunk * 256 + threadIdx.x;
  float cx = sx[n], cy = sy[n], cz = sz[n];

  float bd[KNN];
  int   bi[KNN];
#pragma unroll
  for (int j = 0; j < KNN; ++j) { bd[j] = FLT_MAX; bi[j] = 0; }

  for (int m = 0; m < NPTS; ++m) {
    float dx = cx - sx[m], dy = cy - sy[m], dz = cz - sz[m];
    float d = dx * dx + dy * dy + dz * dz;
    if (d < bd[KNN - 1]) {
      int j = KNN - 1;
      while (j > 0 && d < bd[j - 1]) { bd[j] = bd[j - 1]; bi[j] = bi[j - 1]; --j; }
      bd[j] = d; bi[j] = m;
    }
  }
  int* o = idx + ((size_t)b * NPTS + n) * KNN;
#pragma unroll
  for (int j = 0; j < KNN; ++j) o[j] = bi[j];
}

// ---------------------------------------------------------------------------
// 2) EdgeConv layer 0: edge=[nbr-ctr, ctr] (6ch) -> W0(64x6)+b0 -> lrelu ->
//    max over K. Tiny 6-deep GEMM -> VALU; W0/b0 staged in LDS.
//    Output written into cat[:, 0:64] (row stride 512).
// ---------------------------------------------------------------------------
__global__ void edgeconv0_kernel(const float* __restrict__ x,
                                 const int* __restrict__ idx,
                                 const float* __restrict__ W0,
                                 const float* __restrict__ b0,
                                 float* __restrict__ cat) {
  __shared__ float sW[64 * 6];
  __shared__ float sb[64];
  for (int i = threadIdx.x; i < 64 * 6; i += blockDim.x) sW[i] = W0[i];
  for (int i = threadIdx.x; i < 64; i += blockDim.x) sb[i] = b0[i];
  __syncthreads();

  int m = blockIdx.x * blockDim.x + threadIdx.x;   // global point id
  int b = m / NPTS;
  const float* xb = x + (size_t)b * NPTS * 3;
  float cx = x[(size_t)m * 3 + 0];
  float cy = x[(size_t)m * 3 + 1];
  float cz = x[(size_t)m * 3 + 2];

  float acc[64];
#pragma unroll
  for (int o = 0; o < 64; ++o) acc[o] = -FLT_MAX;

  const int* id = idx + (size_t)m * KNN;
  for (int k = 0; k < KNN; ++k) {
    int j = id[k];
    float e0 = xb[j * 3 + 0] - cx;
    float e1 = xb[j * 3 + 1] - cy;
    float e2 = xb[j * 3 + 2] - cz;
#pragma unroll
    for (int o = 0; o < 64; ++o) {
      const float* w = sW + o * 6;
      float s = sb[o] + w[0] * e0 + w[1] * e1 + w[2] * e2
                      + w[3] * cx + w[4] * cy + w[5] * cz;
      acc[o] = fmaxf(acc[o], lrelu(s));
    }
  }
  float* out = cat + (size_t)m * 512;
#pragma unroll
  for (int o = 0; o < 64; ++o) out[o] = acc[o];
}

// ---------------------------------------------------------------------------
// 3) Graph max-pool: pooled[m,c] = max_k cat[(b,idx[m,k]), coloff+c]
// ---------------------------------------------------------------------------
__global__ void gather_max_kernel(const float* __restrict__ cat,
                                  const int* __restrict__ idx,
                                  float* __restrict__ pooled,
                                  int C, int coloff) {
  int t = blockIdx.x * blockDim.x + threadIdx.x;   // over M*C exactly
  int m = t / C;
  int c = t - m * C;
  int b = m / NPTS;
  const int* id = idx + (size_t)m * KNN;
  size_t rowbase = (size_t)b * NPTS;
  float v = -FLT_MAX;
  for (int k = 0; k < KNN; ++k) {
    int j = id[k];
    v = fmaxf(v, cat[(rowbase + j) * 512 + coloff + c]);
  }
  pooled[(size_t)m * C + c] = v;
}

// ---------------------------------------------------------------------------
// Weight fp32 -> bf16 (RNE) prep
// ---------------------------------------------------------------------------
__global__ void w2bf_kernel(const float* __restrict__ src,
                            unsigned short* __restrict__ dst, int n) {
  int i = blockIdx.x * blockDim.x + threadIdx.x;
  if (i < n) dst[i] = f2bf(src[i]);
}

// ---------------------------------------------------------------------------
// 4) WMMA linear: Y = lrelu(X * W^T + bias). One wave computes a 16x64 tile:
//    4 accumulators, A (fp32->bf16, converted once per K-step) reused across
//    4 v_wmma_f32_16x16x32_bf16; B loaded pre-converted bf16 (2x16B/lane).
//    maxmode=1: fuse row-max + wave32 shfl_xor + float atomicMax into (B,1024).
// ---------------------------------------------------------------------------
__global__ void linear_wmma4_kernel(const float* __restrict__ X, int ldx,
                                    const unsigned short* __restrict__ Wbf,
                                    const float* __restrict__ bias,
                                    float* __restrict__ Y, int ldy, int ycol,
                                    int Cin, int Cout,
                                    float* __restrict__ maxout, int maxmode) {
  int wave = (blockIdx.x * blockDim.x + threadIdx.x) >> 5;
  int lane = threadIdx.x & 31;
  int ng = Cout >> 6;                 // groups of 64 output channels
  int mt = wave / ng;
  int g  = wave - mt * ng;
  int half = lane >> 4;               // K-half select (ISA 16-bit A/B layout)
  int lidx = lane & 15;               // row (A) / column (B)
  int klo  = half * 8;

  const float* arow = X + (size_t)(mt * 16 + lidx) * ldx;

  v8f acc[NT] = {};
  for (int k0 = 0; k0 < Cin; k0 += 32) {
    union { v16bf v; unsigned short u[16]; } A;
    const float4* pa0 = (const float4*)(arow + k0 + klo);
    const float4* pa1 = (const float4*)(arow + k0 + 16 + klo);
    float4 a0 = pa0[0], a1 = pa0[1], a2 = pa1[0], a3 = pa1[1];
    A.u[0] = f2bf(a0.x); A.u[1] = f2bf(a0.y); A.u[2] = f2bf(a0.z); A.u[3] = f2bf(a0.w);
    A.u[4] = f2bf(a1.x); A.u[5] = f2bf(a1.y); A.u[6] = f2bf(a1.z); A.u[7] = f2bf(a1.w);
    A.u[8]  = f2bf(a2.x); A.u[9]  = f2bf(a2.y); A.u[10] = f2bf(a2.z); A.u[11] = f2bf(a2.w);
    A.u[12] = f2bf(a3.x); A.u[13] = f2bf(a3.y); A.u[14] = f2bf(a3.z); A.u[15] = f2bf(a3.w);

#pragma unroll
    for (int t = 0; t < NT; ++t) {
      union { v16bf v; uint4 q[2]; } B;
      const unsigned short* brow = Wbf + (size_t)(g * 64 + t * 16 + lidx) * Cin;
      B.q[0] = *(const uint4*)(brow + k0 + klo);        // bf16 K klo..klo+7
      B.q[1] = *(const uint4*)(brow + k0 + 16 + klo);   // bf16 K 16+klo..+7
      acc[t] = __builtin_amdgcn_wmma_f32_16x16x32_bf16(
          false, A.v, false, B.v, (short)0, acc[t], false, false);
    }
  }

  if (!maxmode) {
#pragma unroll
    for (int t = 0; t < NT; ++t) {
      float bv = bias[g * 64 + t * 16 + lidx];   // D column == lidx
#pragma unroll
      for (int r = 0; r < 8; ++r) {
        int row = mt * 16 + half * 8 + r;
        Y[(size_t)row * ldy + ycol + g * 64 + t * 16 + lidx] =
            lrelu(acc[t][r] + bv);
      }
    }
  } else {
    int b = (mt * 16) / NPTS;                    // 2048 % 16 == 0
#pragma unroll
    for (int t = 0; t < NT; ++t) {
      float bv = bias[g * 64 + t * 16 + lidx];
      float v = -FLT_MAX;
#pragma unroll
      for (int r = 0; r < 8; ++r) v = fmaxf(v, lrelu(acc[t][r] + bv));
      v = fmaxf(v, __shfl_xor(v, 16, 32));       // combine row halves (wave32)
      if (half == 0)
        atomicMax(&maxout[(size_t)b * 1024 + g * 64 + t * 16 + lidx], v);
    }
  }
}

__global__ void init_out_kernel(float* __restrict__ out, int n) {
  int i = blockIdx.x * blockDim.x + threadIdx.x;
  if (i < n) out[i] = -FLT_MAX;
}

// ---------------------------------------------------------------------------
extern "C" void kernel_launch(void* const* d_in, const int* in_sizes, int n_in,
                              void* d_out, int out_size, void* d_ws, size_t ws_size,
                              hipStream_t stream) {
  (void)in_sizes; (void)n_in; (void)out_size; (void)ws_size;
  const float* x  = (const float*)d_in[0];
  const float* W0 = (const float*)d_in[1];
  const float* b0 = (const float*)d_in[2];
  const float* W1 = (const float*)d_in[3];
  const float* b1 = (const float*)d_in[4];
  const float* W2 = (const float*)d_in[5];
  const float* b2 = (const float*)d_in[6];
  const float* W3 = (const float*)d_in[7];
  const float* b3 = (const float*)d_in[8];
  const float* Wf = (const float*)d_in[9];
  const float* bf = (const float*)d_in[10];
  float* out = (float*)d_out;

  const int M = NB * NPTS;                      // 32768 rows
  char* ws = (char*)d_ws;
  int*   idx    = (int*)ws;                               //  2.5 MB
  float* cat    = (float*)(ws + 2621440);                 // 64.0 MB
  float* pooled = (float*)(ws + 2621440 + 67108864);      // 16.0 MB
  unsigned short* W1b = (unsigned short*)(ws + 86507520); //   8 KB
  unsigned short* W2b = (unsigned short*)(ws + 86507520 + 8192);
  unsigned short* W3b = (unsigned short*)(ws + 86507520 + 8192 + 16384);
  unsigned short* Wfb = (unsigned short*)(ws + 86507520 + 8192 + 16384 + 65536);

  // Weight bf16 prep (tiny; overlaps nothing it reads)
  w2bf_kernel<<<(64 * 64) / 256, 256, 0, stream>>>(W1, W1b, 64 * 64);
  w2bf_kernel<<<(128 * 64) / 256, 256, 0, stream>>>(W2, W2b, 128 * 64);
  w2bf_kernel<<<(256 * 128) / 256, 256, 0, stream>>>(W3, W3b, 256 * 128);
  w2bf_kernel<<<(1024 * 512) / 256, 256, 0, stream>>>(Wf, Wfb, 1024 * 512);

  // kNN graph
  knn_kernel<<<NB * 8, 256, 0, stream>>>(x, idx);
  // EdgeConv -> cat[:, 0:64]
  edgeconv0_kernel<<<M / 128, 128, 0, stream>>>(x, idx, W0, b0, cat);
  // Layer 1: pool(feat0) -> 64->64 -> cat[:, 64:128]
  gather_max_kernel<<<(M * 64) / 256, 256, 0, stream>>>(cat, idx, pooled, 64, 0);
  linear_wmma4_kernel<<<(M / 16) * (64 / 64) / 8, 256, 0, stream>>>(
      pooled, 64, W1b, b1, cat, 512, 64, 64, 64, nullptr, 0);
  // Layer 2: pool(feat1) -> 64->128 -> cat[:, 128:256]
  gather_max_kernel<<<(M * 64) / 256, 256, 0, stream>>>(cat, idx, pooled, 64, 64);
  linear_wmma4_kernel<<<(M / 16) * (128 / 64) / 8, 256, 0, stream>>>(
      pooled, 64, W2b, b2, cat, 512, 128, 64, 128, nullptr, 0);
  // Layer 3: pool(feat2) -> 128->256 -> cat[:, 256:512]
  gather_max_kernel<<<(M * 128) / 256, 256, 0, stream>>>(cat, idx, pooled, 128, 128);
  linear_wmma4_kernel<<<(M / 16) * (256 / 64) / 8, 256, 0, stream>>>(
      pooled, 128, W3b, b3, cat, 512, 256, 128, 256, nullptr, 0);
  // Final: 512 -> 1024, fused global max over N via atomicMax
  init_out_kernel<<<(NB * 1024) / 256, 256, 0, stream>>>(out, NB * 1024);
  linear_wmma4_kernel<<<(M / 16) * (1024 / 64) / 8, 256, 0, stream>>>(
      cat, 512, Wfb, bf, nullptr, 0, 0, 512, 1024, out, 1);
}
